// SparseMoE_609885356203
// MI455X (gfx1250) — matrix operand order repaired and checked by reading, not compiled, not run on hardware
//
#include <hip/hip_runtime.h>
#include <hip/hip_bf16.h>

// Problem constants (D, E, K, H = 1024, 8, 2, 4096; B,S = 2,2048)
#define D_ 1024
#define E_ 8
#define T_ 4096   // B*S
#define H_ 4096

typedef __attribute__((ext_vector_type(16))) __bf16 v16bf;
typedef __attribute__((ext_vector_type(8)))  float  v8f;

__device__ __forceinline__ __bf16 f2bf(float f) { return (__bf16)f; }

__device__ __forceinline__ float gelu_exact(float v) {
  return 0.5f * v * (1.0f + erff(v * 0.70710678118654752440f));
}

__device__ __forceinline__ float softplus_f(float z) {
  return (z > 0.f) ? (z + log1pf(expf(-z))) : log1pf(expf(z));
}

// 16-bit A-matrix 16x32 layout: lane m=lane&15, hi=lane>>4.
// VGPR r holds K pair at base = hi*16 + 2*r (contiguous, matches B packing).
__device__ __forceinline__ v16bf load_a_frag(const __bf16* row, int hi) {
  v16bf a;
#pragma unroll
  for (int r = 0; r < 8; ++r) {
    int kb = (hi << 4) + (r << 1);
    a[2 * r]     = row[kb];
    a[2 * r + 1] = row[kb + 1];
  }
  return a;
}

__global__ void moe_zero(float* __restrict__ out, int n, int* __restrict__ cnt) {
  int i = blockIdx.x * blockDim.x + threadIdx.x;
  if (i < n) out[i] = 0.f;
  if (i < E_) cnt[i] = 0;
}

__global__ void moe_router(const float* __restrict__ x, const float* __restrict__ eps,
                           const float* __restrict__ Wr, const float* __restrict__ br,
                           const float* __restrict__ Wn, const float* __restrict__ bn,
                           int* __restrict__ cnt, int* __restrict__ list,
                           float* __restrict__ topg) {
  const int lane = threadIdx.x & 31;
  const int wv   = threadIdx.x >> 5;
  const int t    = blockIdx.x * 8 + wv;      // one wave32 per token
  if (t >= T_) return;

  float aR[E_], aN[E_];
#pragma unroll
  for (int e = 0; e < E_; ++e) { aR[e] = 0.f; aN[e] = 0.f; }

  const float* xr = x + (size_t)t * D_;
  for (int d = lane; d < D_; d += 32) {
    float xv = xr[d];
    const float4* wr4 = reinterpret_cast<const float4*>(Wr + d * E_);
    const float4* wn4 = reinterpret_cast<const float4*>(Wn + d * E_);
    float4 r0 = wr4[0], r1 = wr4[1], n0 = wn4[0], n1 = wn4[1];
    aR[0] += xv * r0.x; aR[1] += xv * r0.y; aR[2] += xv * r0.z; aR[3] += xv * r0.w;
    aR[4] += xv * r1.x; aR[5] += xv * r1.y; aR[6] += xv * r1.z; aR[7] += xv * r1.w;
    aN[0] += xv * n0.x; aN[1] += xv * n0.y; aN[2] += xv * n0.z; aN[3] += xv * n0.w;
    aN[4] += xv * n1.x; aN[5] += xv * n1.y; aN[6] += xv * n1.z; aN[7] += xv * n1.w;
  }
#pragma unroll
  for (int e = 0; e < E_; ++e) {
#pragma unroll
    for (int off = 16; off > 0; off >>= 1) {
      aR[e] += __shfl_xor(aR[e], off, 32);
      aN[e] += __shfl_xor(aN[e], off, 32);
    }
  }

  if (lane == 0) {
    float v[E_];
#pragma unroll
    for (int e = 0; e < E_; ++e) {
      float lg = aR[e] + br[e];
      float nl = aN[e] + bn[e];
      v[e] = lg + eps[t * E_ + e] * softplus_f(nl);
    }
    int i1 = 0; float v1 = v[0];
#pragma unroll
    for (int e = 1; e < E_; ++e) if (v[e] > v1) { v1 = v[e]; i1 = e; }
    int i2 = (i1 == 0) ? 1 : 0; float v2 = v[i2];
#pragma unroll
    for (int e = 0; e < E_; ++e)
      if (e != i1 && e != ((i1 == 0) ? 1 : 0) && v[e] > v2) { v2 = v[e]; i2 = e; }
    float s  = expf(v2 - v1);
    float g1 = 1.f / (1.f + s);
    float g2 = s / (1.f + s);
    topg[t * 2 + 0] = g1;
    topg[t * 2 + 1] = g2;
    int p1 = atomicAdd(&cnt[i1], 1);
    list[i1 * T_ + p1] = t * 2;
    int p2 = atomicAdd(&cnt[i2], 1);
    list[i2 * T_ + p2] = t * 2 + 1;
  }
}

// Pack fp32 weights [E][Kdim][Ndim] -> bf16 WMMA B-fragments:
// P[((e*Ntiles + nt)*Ksteps + ks)*512 + lane*16 + j], element j of lane holds
// B[k0 + (lane>>4)*16 + j][nt*16 + (lane&15)].  One contiguous 1KB per fragment.
__global__ __launch_bounds__(256) void moe_pack(const float* __restrict__ W,
                                                __bf16* __restrict__ P,
                                                int Ksteps, int Ntiles) {
  __shared__ __bf16 Ls[32 * 128];
  const int ks = blockIdx.x;
  const int cb = blockIdx.y;            // 128-column block
  const int e  = blockIdx.z;
  const int Ndim = Ntiles * 16;
  const float* We = W + (size_t)e * (size_t)(Ksteps * 32) * Ndim;
  const int tid = threadIdx.x;
  {
    int jj = tid & 127, kkb = (tid >> 7) << 4;
    int k0 = ks * 32, c0 = cb * 128;
#pragma unroll 4
    for (int it = 0; it < 16; ++it) {
      int kk = kkb + it;
      Ls[kk * 128 + jj] = f2bf(We[(size_t)(k0 + kk) * Ndim + c0 + jj]);
    }
  }
  __syncthreads();
  const int ntl = tid >> 5, lane = tid & 31;
  const int n = lane & 15, khalf = (lane >> 4) << 4;
  __bf16* dst = P + ((size_t)(e * Ntiles + cb * 8 + ntl) * Ksteps + ks) * 512 + lane * 16;
#pragma unroll
  for (int j = 0; j < 16; ++j) dst[j] = Ls[(khalf + j) * 128 + ntl * 16 + n];
}

// ---------------- Fast path: B fragments loaded straight from packed global ----
__global__ __launch_bounds__(256) void moe_expert_packed(
    const float* __restrict__ x,
    const __bf16* __restrict__ P1, const float* __restrict__ b1,
    const __bf16* __restrict__ P2, const float* __restrict__ b2,
    const int* __restrict__ cnt, const int* __restrict__ list,
    const float* __restrict__ topg, float* __restrict__ out) {
  __shared__ __align__(32) __bf16 Xs[16 * D_];   // 32 KB token tile (bf16)
  __shared__ __align__(32) __bf16 Hs[16 * 128];  //  4 KB gelu(h) chunk
  __shared__ int ent[16];

  const int e   = blockIdx.y;
  const int n_e = cnt[e];
  const int r0  = blockIdx.x * 16;
  if (r0 >= n_e) return;

  const int tid  = threadIdx.x;
  const int lane = tid & 31;
  const int wv   = tid >> 5;
  const int m    = lane & 15;
  const int hi   = lane >> 4;

  if (tid < 16) {
    int rr = r0 + tid;
    ent[tid] = (rr < n_e) ? list[e * T_ + rr] : -1;
  }
  __syncthreads();

  {   // gather X rows -> bf16 LDS
    int rr = tid >> 4, cc = tid & 15;
    int ev = ent[rr];
    if (ev >= 0) {
      const float* xr = x + (size_t)(ev >> 1) * D_;
#pragma unroll 4
      for (int it = 0; it < 16; ++it) {
        int d0 = cc * 64 + it * 4;
        float4 v = *reinterpret_cast<const float4*>(xr + d0);
        Xs[rr * D_ + d0 + 0] = f2bf(v.x);
        Xs[rr * D_ + d0 + 1] = f2bf(v.y);
        Xs[rr * D_ + d0 + 2] = f2bf(v.z);
        Xs[rr * D_ + d0 + 3] = f2bf(v.w);
      }
    } else {
#pragma unroll 4
      for (int it = 0; it < 16; ++it) {
        int d0 = cc * 64 + it * 4;
        Xs[rr * D_ + d0 + 0] = f2bf(0.f);
        Xs[rr * D_ + d0 + 1] = f2bf(0.f);
        Xs[rr * D_ + d0 + 2] = f2bf(0.f);
        Xs[rr * D_ + d0 + 3] = f2bf(0.f);
      }
    }
  }
  __syncthreads();

  const __bf16* P1e = P1 + (size_t)e * (size_t)(H_ / 16) * (D_ / 32) * 512;
  const __bf16* P2e = P2 + (size_t)e * (size_t)(D_ / 16) * (H_ / 32) * 512;
  const float*  b1e = b1 + (size_t)e * H_;
  const float*  b2e = b2 + (size_t)e * D_;

  v8f yacc[8];
#pragma unroll
  for (int i = 0; i < 8; ++i) yacc[i] = (v8f){0.f, 0.f, 0.f, 0.f, 0.f, 0.f, 0.f, 0.f};

  for (int ch = 0; ch < H_ / 128; ++ch) {
    const int h0 = ch * 128;

    // GEMM1: wave computes h tile columns [h0+wv*16, +16), K = D
    v8f acc = (v8f){0.f, 0.f, 0.f, 0.f, 0.f, 0.f, 0.f, 0.f};
    const __bf16* fb1 = P1e + (size_t)(ch * 8 + wv) * (D_ / 32) * 512 + lane * 16;
#pragma unroll 4
    for (int ks = 0; ks < D_ / 32; ++ks) {
      __builtin_prefetch(fb1 + (((ks + 4) & 31) * 512), 0, 0);  // wraps in-region
      v16bf a = load_a_frag(Xs + m * D_ + ks * 32, hi);
      v16bf b = *reinterpret_cast<const v16bf*>(fb1 + (size_t)ks * 512);
      acc = __builtin_amdgcn_wmma_f32_16x16x32_bf16(false, a, false, b,
                                                    (short)0, acc, false, false);
    }
#pragma unroll
    for (int r = 0; r < 8; ++r) {
      int mm   = hi * 8 + r;
      int colg = h0 + wv * 16 + m;
      Hs[mm * 128 + wv * 16 + m] = f2bf(gelu_exact(acc[r] + b1e[colg]));
    }
    __syncthreads();

    // GEMM2: y[16,1024] += h_chunk[16,128] @ W2[h0:h0+128, :]
    v16bf ah[4];
#pragma unroll
    for (int ks = 0; ks < 4; ++ks) ah[ks] = load_a_frag(Hs + m * 128 + ks * 32, hi);
#pragma unroll
    for (int i = 0; i < 8; ++i) {
      int nt = wv * 8 + i;
      const __bf16* fb2 = P2e + ((size_t)nt * (H_ / 32) + ch * 4) * 512 + lane * 16;
#pragma unroll
      for (int ks = 0; ks < 4; ++ks) {
        v16bf b = *reinterpret_cast<const v16bf*>(fb2 + (size_t)ks * 512);
        yacc[i] = __builtin_amdgcn_wmma_f32_16x16x32_bf16(
            false, ah[ks], false, b, (short)0, yacc[i], false, false);
      }
    }
    __syncthreads();
  }

#pragma unroll
  for (int i = 0; i < 8; ++i) {
    int col = (wv * 8 + i) * 16 + m;
    float bb = b2e[col];
#pragma unroll
    for (int r = 0; r < 8; ++r) {
      int ev = ent[hi * 8 + r];
      if (ev >= 0) {
        float g = topg[ev];
        atomicAdd(out + (size_t)(ev >> 1) * D_ + col, g * (yacc[i][r] + bb));
      }
    }
  }
}

// ---------------- Fallback path (small ws): convert weights in-kernel ----------
__global__ __launch_bounds__(256) void moe_expert_direct(
    const float* __restrict__ x,
    const float* __restrict__ W1, const float* __restrict__ b1,
    const float* __restrict__ W2, const float* __restrict__ b2,
    const int* __restrict__ cnt, const int* __restrict__ list,
    const float* __restrict__ topg, float* __restrict__ out) {
  __shared__ __align__(32) __bf16 Xs[16 * D_];
  __shared__ __align__(32) __bf16 Hs[16 * 128];
  __shared__ __align__(32) __bf16 Bs[32 * 256];
  __shared__ int ent[16];

  const int e   = blockIdx.y;
  const int n_e = cnt[e];
  const int r0  = blockIdx.x * 16;
  if (r0 >= n_e) return;

  const int tid  = threadIdx.x;
  const int lane = tid & 31;
  const int wv   = tid >> 5;
  const int m    = lane & 15;
  const int hi   = lane >> 4;

  if (tid < 16) {
    int rr = r0 + tid;
    ent[tid] = (rr < n_e) ? list[e * T_ + rr] : -1;
  }
  __syncthreads();

  {
    int rr = tid >> 4, cc = tid & 15;
    int ev = ent[rr];
    if (ev >= 0) {
      const float* xr = x + (size_t)(ev >> 1) * D_;
#pragma unroll 4
      for (int it = 0; it < 16; ++it) {
        int d0 = cc * 64 + it * 4;
        float4 v = *reinterpret_cast<const float4*>(xr + d0);
        Xs[rr * D_ + d0 + 0] = f2bf(v.x);
        Xs[rr * D_ + d0 + 1] = f2bf(v.y);
        Xs[rr * D_ + d0 + 2] = f2bf(v.z);
        Xs[rr * D_ + d0 + 3] = f2bf(v.w);
      }
    } else {
#pragma unroll 4
      for (int it = 0; it < 16; ++it) {
        int d0 = cc * 64 + it * 4;
        Xs[rr * D_ + d0 + 0] = f2bf(0.f);
        Xs[rr * D_ + d0 + 1] = f2bf(0.f);
        Xs[rr * D_ + d0 + 2] = f2bf(0.f);
        Xs[rr * D_ + d0 + 3] = f2bf(0.f);
      }
    }
  }
  __syncthreads();

  const float* W1e = W1 + (size_t)e * D_ * H_;
  const float* W2e = W2 + (size_t)e * H_ * D_;
  const float* b1e = b1 + (size_t)e * H_;
  const float* b2e = b2 + (size_t)e * D_;

  v8f yacc[8];
#pragma unroll
  for (int i = 0; i < 8; ++i) yacc[i] = (v8f){0.f, 0.f, 0.f, 0.f, 0.f, 0.f, 0.f, 0.f};

  for (int ch = 0; ch < H_ / 128; ++ch) {
    const int h0 = ch * 128;
    v8f acc = (v8f){0.f, 0.f, 0.f, 0.f, 0.f, 0.f, 0.f, 0.f};
    for (int ks = 0; ks < D_ / 32; ++ks) {
      const int k0 = ks * 32;
      {
        int jj  = tid & 127;
        int kkb = (tid >> 7) << 4;
#pragma unroll 4
        for (int it = 0; it < 16; ++it) {
          int kk = kkb + it;
          float v = W1e[(size_t)(k0 + kk) * H_ + (h0 + jj)];
          Bs[(((jj >> 4) * 32) + ((kk >> 4) << 4) + (jj & 15)) * 16 + (kk & 15)] = f2bf(v);
        }
      }
      __syncthreads();
      v16bf a = load_a_frag(Xs + m * D_ + k0, hi);
      v16bf b = *reinterpret_cast<const v16bf*>(Bs + (wv * 32 + lane) * 16);
      acc = __builtin_amdgcn_wmma_f32_16x16x32_bf16(false, a, false, b,
                                                    (short)0, acc, false, false);
      __syncthreads();
    }
#pragma unroll
    for (int r = 0; r < 8; ++r) {
      int mm   = hi * 8 + r;
      int colg = h0 + wv * 16 + m;
      Hs[mm * 128 + wv * 16 + m] = f2bf(gelu_exact(acc[r] + b1e[colg]));
    }
    __syncthreads();

    for (int ks = 0; ks < 4; ++ks) {
      const int kh = ks * 32;
      for (int sub = 0; sub < 4; ++sub) {
        {
          int jj = tid;
#pragma unroll 4
          for (int kk = 0; kk < 32; ++kk) {
            float v = W2e[(size_t)(h0 + kh + kk) * D_ + sub * 256 + jj];
            Bs[(((jj >> 4) * 32) + ((kk >> 4) << 4) + (jj & 15)) * 16 + (kk & 15)] = f2bf(v);
          }
        }
        __syncthreads();
        v16bf a = load_a_frag(Hs + m * 128 + kh, hi);
#pragma unroll
        for (int j = 0; j < 2; ++j) {
          int ntl = wv * 2 + j;
          v16bf b = *reinterpret_cast<const v16bf*>(Bs + (ntl * 32 + lane) * 16);
          yacc[sub * 2 + j] = __builtin_amdgcn_wmma_f32_16x16x32_bf16(
              false, a, false, b, (short)0, yacc[sub * 2 + j], false, false);
        }
        __syncthreads();
      }
    }
  }

#pragma unroll
  for (int i = 0; i < 8; ++i) {
    int sub = i >> 1, j = i & 1;
    int col = sub * 256 + (wv * 2 + j) * 16 + m;
    float bb = b2e[col];
#pragma unroll
    for (int r = 0; r < 8; ++r) {
      int ev = ent[hi * 8 + r];
      if (ev >= 0) {
        float g = topg[ev];
        atomicAdd(out + (size_t)(ev >> 1) * D_ + col, g * (yacc[i][r] + bb));
      }
    }
  }
}

extern "C" void kernel_launch(void* const* d_in, const int* in_sizes, int n_in,
                              void* d_out, int out_size, void* d_ws, size_t ws_size,
                              hipStream_t stream) {
  const float* x   = (const float*)d_in[0];
  const float* eps = (const float*)d_in[1];
  const float* Wr  = (const float*)d_in[2];
  const float* br  = (const float*)d_in[3];
  const float* Wn  = (const float*)d_in[4];
  const float* bn  = (const float*)d_in[5];
  const float* W1  = (const float*)d_in[6];
  const float* b1  = (const float*)d_in[7];
  const float* W2  = (const float*)d_in[8];
  const float* b2  = (const float*)d_in[9];
  float* out = (float*)d_out;

  char*  ws   = (char*)d_ws;
  int*   cnt  = (int*)ws;                              // E ints
  int*   list = (int*)(ws + 256);                      // E*T ints (128 KB)
  float* topg = (float*)(ws + 256 + E_ * T_ * 4);      // T*2 floats (32 KB)
  const size_t ctrlBytes = 256 + (size_t)E_ * T_ * 4 + (size_t)T_ * 2 * 4;  // 164096
  const size_t packElems = (size_t)E_ * D_ * H_;       // per weight tensor
  const size_t needBytes = ctrlBytes + 2 * packElems * sizeof(__bf16);      // ~128.2 MB

  const int n = T_ * D_;
  moe_zero<<<(n + 255) / 256, 256, 0, stream>>>(out, n, cnt);
  moe_router<<<T_ / 8, 256, 0, stream>>>(x, eps, Wr, br, Wn, bn, cnt, list, topg);

  dim3 grid(T_ / 16, E_);
  if (ws_size >= needBytes) {
    __bf16* P1 = (__bf16*)(ws + ctrlBytes);
    __bf16* P2 = P1 + packElems;
    dim3 g1(D_ / 32, H_ / 128, E_);   // K-steps, 128-col blocks, experts
    dim3 g2(H_ / 32, D_ / 128, E_);
    moe_pack<<<g1, 256, 0, stream>>>(W1, P1, D_ / 32, H_ / 16);
    moe_pack<<<g2, 256, 0, stream>>>(W2, P2, H_ / 32, D_ / 16);
    moe_expert_packed<<<grid, 256, 0, stream>>>(x, P1, b1, P2, b2, cnt, list, topg, out);
  } else {
    moe_expert_direct<<<grid, 256, 0, stream>>>(x, W1, b1, W2, b2, cnt, list, topg, out);
  }
}
